// MHA2_49538152792560
// MI455X (gfx1250) — matrix operand order repaired and checked by reading, not compiled, not run on hardware
//
#include <hip/hip_runtime.h>

typedef __attribute__((ext_vector_type(16))) __bf16 v16bf;
typedef __attribute__((ext_vector_type(8)))  float  v8f;
typedef __attribute__((ext_vector_type(4)))  unsigned int u32x4;
typedef __attribute__((ext_vector_type(4)))  int i32x4;
typedef __attribute__((ext_vector_type(8)))  int i32x8;

#define DIN    1024
#define DIM    1024
#define NHEADS 16
#define HDIM   64
#define SEQ    2048
#define BATCH  2
#define MROWS  (BATCH*SEQ)   // 4096

union FragBF { v16bf v; u32x4 q[2]; };
union AccF   { v8f v; float f[8]; };

__device__ __forceinline__ unsigned short f2bf(float f) {
  unsigned int u = __float_as_uint(f);
  u += 0x7FFFu + ((u >> 16) & 1u);   // round-to-nearest-even
  return (unsigned short)(u >> 16);
}

__device__ __forceinline__ unsigned lds_off(const void* p) {
  // Generic pointer to __shared__: low 32 bits are the wave-relative LDS offset.
  return (unsigned)(unsigned long long)p;
}

// Issue one TDM 2D tile load: tile_x elems per row (2B each), tile_y rows,
// global row stride = row_len elems.  LDS padding encoded per ISA 8.4:
// pad_interval: bytes-per-row = 8<<pi ; pad_amount: (pa+1) DWORDs appended.
__device__ __forceinline__ void tdm_load_2d(
    unsigned lds_byte_off, const unsigned short* gptr,
    unsigned tile_x, unsigned tile_y, unsigned row_len,
    unsigned pad_interval, unsigned pad_amount) {
  unsigned long long ga = (unsigned long long)gptr;
  u32x4 g0;
  g0[0] = 1u;                                          // count=1 (valid user D#)
  g0[1] = lds_byte_off;                                // lds_addr
  g0[2] = (unsigned)(ga & 0xFFFFFFFFu);                // global_addr[31:0]
  g0[3] = (unsigned)((ga >> 32) & 0x01FFFFFFu)         // global_addr[56:32]
        | (2u << 30);                                  // type = 2 ("image")
  unsigned w0 = (1u << 16)                             // data_size = 2 bytes
              | (1u << 20)                             // pad_enable
              | (pad_interval << 22)
              | (pad_amount << 25);
  unsigned long long stride0 = row_len;                // tensor_dim0_stride
  unsigned long long g1a = (unsigned long long)w0
      | ((unsigned long long)(row_len & 0xFFFFu) << 48);        // tensor_dim0 lo
  unsigned long long g1b = ((unsigned long long)row_len >> 16)   // tensor_dim0 hi
      | ((unsigned long long)tile_y << 16)                       // tensor_dim1
      | ((unsigned long long)tile_x << 48);                      // tile_dim0
  unsigned long long g1c = (unsigned long long)tile_y            // tile_dim1
      | ((stride0 & 0xFFFFFFFFull) << 32);                       // stride0 lo
  unsigned long long g1d = (stride0 >> 32) & 0xFFFFull;          // stride0 hi
  i32x8 g1;
  g1[0] = (int)(unsigned)g1a; g1[1] = (int)(g1a >> 32);
  g1[2] = (int)(unsigned)g1b; g1[3] = (int)(g1b >> 32);
  g1[4] = (int)(unsigned)g1c; g1[5] = (int)(g1c >> 32);
  g1[6] = (int)(unsigned)g1d; g1[7] = (int)(g1d >> 32);
  i32x4 z4 = {0, 0, 0, 0};
#if __clang_major__ >= 23
  i32x8 z8 = {0, 0, 0, 0, 0, 0, 0, 0};
  __builtin_amdgcn_tensor_load_to_lds(g0, g1, z4, z4, z8, 0);
#else
  __builtin_amdgcn_tensor_load_to_lds(g0, g1, z4, z4, 0);
#endif
}

__global__ void cvt_f32_bf16(const float* __restrict__ src,
                             unsigned short* __restrict__ dst, int n) {
  int i = blockIdx.x * blockDim.x + threadIdx.x;
  int stride = gridDim.x * blockDim.x;
  for (; i < n; i += stride) dst[i] = f2bf(src[i]);
}

// C = A[M,K] * W[N,K]^T (bf16 in, f32 accumulate).  TDM double-buffered tiles.
template <bool OUTF32>
__global__ __launch_bounds__(256) void gemm_bf16_nt(
    const unsigned short* __restrict__ A,
    const unsigned short* __restrict__ W,
    float* __restrict__ Cf, unsigned short* __restrict__ Cb,
    int M, int N, int K) {
  constexpr int LDSS = 40;  // 80B rows (64B data + 16B TDM pad) -> conflict free
  __shared__ __align__(16) unsigned short At[2][128 * LDSS];
  __shared__ __align__(16) unsigned short Wt[2][128 * LDSS];

  const int tid  = threadIdx.x;
  const int lane = tid & 31;
  const int wid  = tid >> 5;     // 8 waves
  const int wm   = wid >> 1;     // 0..3 : 32-row slice
  const int wn   = wid & 1;      // 0..1 : 64-col slice
  const int h    = lane >> 4;    // wave half
  const int ln16 = lane & 15;

  const int bm = blockIdx.y * 128;
  const int bn = blockIdx.x * 128;

  AccF acc[2][4];
#pragma unroll
  for (int i = 0; i < 2; ++i)
#pragma unroll
    for (int j = 0; j < 4; ++j)
#pragma unroll
      for (int e = 0; e < 8; ++e) acc[i][j].f[e] = 0.0f;

  const int nsteps = K / 32;
  if (wid == 0) {  // prologue: DMA first tiles into buffer 0
    tdm_load_2d(lds_off(&At[0][0]), &A[(size_t)bm * K], 32, 128, K, 3, 3);
    tdm_load_2d(lds_off(&Wt[0][0]), &W[(size_t)bn * K], 32, 128, K, 3, 3);
  }

  for (int i = 0; i < nsteps; ++i) {
    const int buf = i & 1;
    if (wid == 0) __builtin_amdgcn_s_wait_tensorcnt(0);  // current buf landed
    __syncthreads();
    if (wid == 0 && i + 1 < nsteps) {  // prefetch next tile into other buffer
      tdm_load_2d(lds_off(&At[buf ^ 1][0]),
                  &A[(size_t)bm * K + (i + 1) * 32], 32, 128, K, 3, 3);
      tdm_load_2d(lds_off(&Wt[buf ^ 1][0]),
                  &W[(size_t)bn * K + (i + 1) * 32], 32, 128, K, 3, 3);
    }

    // A fragments: lane holds row ln16; K chunks at h*8 and h*8+16 (ISA 7.12.2)
    FragBF a[2];
#pragma unroll
    for (int mt = 0; mt < 2; ++mt) {
      int r = wm * 32 + mt * 16 + ln16;
      a[mt].q[0] = *(const u32x4*)&At[buf][r * LDSS + h * 8];
      a[mt].q[1] = *(const u32x4*)&At[buf][r * LDSS + h * 8 + 16];
    }
#pragma unroll
    for (int nt = 0; nt < 4; ++nt) {
      // B fragment: lane = column n = W row n; K chunks at h*8 and h*8+16
      FragBF b;
      int n = wn * 64 + nt * 16 + ln16;
      b.q[0] = *(const u32x4*)&Wt[buf][n * LDSS + h * 8];
      b.q[1] = *(const u32x4*)&Wt[buf][n * LDSS + h * 8 + 16];
#pragma unroll
      for (int mt = 0; mt < 2; ++mt) {
        acc[mt][nt].v = __builtin_amdgcn_wmma_f32_16x16x32_bf16(
            false, a[mt].v, false, b.v, (short)0, acc[mt][nt].v, false, false);
      }
    }
  }

  // C layout: VGPR e -> row h*8+e, lane%16 -> col
#pragma unroll
  for (int mt = 0; mt < 2; ++mt)
#pragma unroll
    for (int nt = 0; nt < 4; ++nt)
#pragma unroll
      for (int e = 0; e < 8; ++e) {
        int r = bm + wm * 32 + mt * 16 + h * 8 + e;
        int c = bn + wn * 64 + nt * 16 + ln16;
        if (OUTF32) Cf[(size_t)r * N + c] = acc[mt][nt].f[e];
        else        Cb[(size_t)r * N + c] = f2bf(acc[mt][nt].f[e]);
      }
}

// Causal flash attention: block = 64 queries of one (batch, head); 4 waves,
// each wave owns 16 query rows.  K tile DMA'd by TDM; V transposed by hand.
__global__ __launch_bounds__(128) void flash_attn_causal(
    const unsigned short* __restrict__ Q,
    const unsigned short* __restrict__ Kg,
    const unsigned short* __restrict__ V,
    unsigned short* __restrict__ ctx) {
  constexpr int LDK = 72;  // 144B rows (128B data + 16B TDM pad)
  constexpr int LDV = 40;
  constexpr int LDP = 40;
  __shared__ __align__(16) unsigned short Klds[32 * LDK];      // [key][d]
  __shared__ __align__(16) unsigned short Vt[64 * LDV];        // [d][key]
  __shared__ __align__(16) unsigned short Plds[4][16 * LDP];   // per-wave P

  const int tid  = threadIdx.x;
  const int lane = tid & 31;
  const int w    = tid >> 5;
  const int h    = lane >> 4;
  const int ln16 = lane & 15;

  const int qb   = blockIdx.x;
  const int head = blockIdx.y;
  const int b    = blockIdx.z;

  const int    qbase   = qb * 64 + w * 16;
  const size_t rowbase = (size_t)b * SEQ * DIM + (size_t)head * HDIM;

  // Q fragments (A layout), resident across the whole key loop.
  FragBF qa[2];
  {
    const unsigned short* qrow = Q + rowbase + (size_t)(qbase + ln16) * DIM;
#pragma unroll
    for (int f = 0; f < 2; ++f) {
      qa[f].q[0] = *(const u32x4*)&qrow[f * 32 + h * 8];
      qa[f].q[1] = *(const u32x4*)&qrow[f * 32 + h * 8 + 16];
    }
  }

  AccF o[4];
#pragma unroll
  for (int t = 0; t < 4; ++t)
#pragma unroll
    for (int e = 0; e < 8; ++e) o[t].f[e] = 0.0f;
  float mrun[8], lrun[8];
#pragma unroll
  for (int i = 0; i < 8; ++i) { mrun[i] = -1e30f; lrun[i] = 0.0f; }

  const float scale  = 0.125f;        // 1/sqrt(HDIM)
  const int   ntiles = (qb + 1) * 2;  // causal bound

  for (int kt = 0; kt < ntiles; ++kt) {
    const int k0 = kt * 32;
    __syncthreads();
    // K tile [32 keys][64 d] via Tensor Data Mover (pads rows to 144B).
    if (w == 0)
      tdm_load_2d(lds_off(&Klds[0]), &Kg[rowbase + (size_t)k0 * DIM],
                  64, 32, DIM, 4, 3);
    // V tile transposed by the wave pipe meanwhile: Vt[d][key].
    for (int idx = tid; idx < 2048; idx += 128) {
      int key = idx >> 6, d = idx & 63;
      Vt[d * LDV + key] = V[rowbase + (size_t)(k0 + key) * DIM + d];
    }
    if (w == 0) __builtin_amdgcn_s_wait_tensorcnt(0);
    __syncthreads();

    // S = Q K^T : two 16x16 key tiles, contraction over d=64 (2 WMMAs each).
    AccF s[2];
#pragma unroll
    for (int t = 0; t < 2; ++t) {
#pragma unroll
      for (int e = 0; e < 8; ++e) s[t].f[e] = 0.0f;
#pragma unroll
      for (int ks = 0; ks < 2; ++ks) {
        FragBF bfr;
        int row = t * 16 + ln16;
        bfr.q[0] = *(const u32x4*)&Klds[row * LDK + ks * 32 + h * 8];
        bfr.q[1] = *(const u32x4*)&Klds[row * LDK + ks * 32 + h * 8 + 16];
        s[t].v = __builtin_amdgcn_wmma_f32_16x16x32_bf16(
            false, qa[ks].v, false, bfr.v, (short)0, s[t].v, false, false);
      }
    }

    // Only diagonal-adjacent tiles of this wave can be masked (wave-uniform).
    const bool needMask = (k0 + 31) > qbase;

    // Online softmax; row r = h*8+e lives in one 16-lane half.
#pragma unroll
    for (int e = 0; e < 8; ++e) {
      int   qr = qbase + h * 8 + e;
      float s0 = s[0].f[e] * scale;
      float s1 = s[1].f[e] * scale;
      if (needMask) {
        if (k0 + ln16 > qr)      s0 = -1e30f;
        if (k0 + 16 + ln16 > qr) s1 = -1e30f;
      }
      float mx = fmaxf(s0, s1);
#pragma unroll
      for (int d = 1; d < 16; d <<= 1) mx = fmaxf(mx, __shfl_xor(mx, d, 32));
      float mnew = fmaxf(mrun[e], mx);
      float corr = __expf(mrun[e] - mnew);
      float p0 = __expf(s0 - mnew);
      float p1 = __expf(s1 - mnew);
      float sum = p0 + p1;
#pragma unroll
      for (int d = 1; d < 16; d <<= 1) sum += __shfl_xor(sum, d, 32);
      lrun[e] = lrun[e] * corr + sum;
      mrun[e] = mnew;
#pragma unroll
      for (int t = 0; t < 4; ++t) o[t].f[e] *= corr;
      Plds[w][(h * 8 + e) * LDP + ln16]      = f2bf(p0);
      Plds[w][(h * 8 + e) * LDP + 16 + ln16] = f2bf(p1);
    }
    __syncthreads();   // C-layout -> A-layout round trip through LDS

    // O += P V : A = P (16x32), B = Vt d-tiles (4 WMMAs).
    FragBF pa;
    {
      const unsigned short* prow = &Plds[w][ln16 * LDP];
      pa.q[0] = *(const u32x4*)&prow[h * 8];
      pa.q[1] = *(const u32x4*)&prow[h * 8 + 16];
    }
#pragma unroll
    for (int t = 0; t < 4; ++t) {
      FragBF bfr;
      int d = t * 16 + ln16;
      bfr.q[0] = *(const u32x4*)&Vt[d * LDV + h * 8];
      bfr.q[1] = *(const u32x4*)&Vt[d * LDV + h * 8 + 16];
      o[t].v = __builtin_amdgcn_wmma_f32_16x16x32_bf16(
          false, pa.v, false, bfr.v, (short)0, o[t].v, false, false);
    }
  }

  // Normalize and emit ctx in [b, s, h*64+d] order (bf16 for the wo GEMM).
#pragma unroll
  for (int e = 0; e < 8; ++e) {
    float inv = 1.0f / lrun[e];
    int   qr  = qbase + h * 8 + e;
    unsigned short* crow = ctx + rowbase + (size_t)qr * DIM;
#pragma unroll
    for (int t = 0; t < 4; ++t)
      crow[t * 16 + ln16] = f2bf(o[t].f[e] * inv);
  }
}

extern "C" void kernel_launch(void* const* d_in, const int* in_sizes, int n_in,
                              void* d_out, int out_size, void* d_ws, size_t ws_size,
                              hipStream_t stream) {
  const float* x  = (const float*)d_in[0];
  const float* wq = (const float*)d_in[1];
  const float* wk = (const float*)d_in[2];
  const float* wv = (const float*)d_in[3];
  const float* wo = (const float*)d_in[4];
  float* out = (float*)d_out;

  unsigned short* ws  = (unsigned short*)d_ws;
  unsigned short* xb  = ws;
  unsigned short* wqb = xb  + (size_t)MROWS * DIN;
  unsigned short* wkb = wqb + (size_t)DIM * DIN;
  unsigned short* wvb = wkb + (size_t)DIM * DIN;
  unsigned short* wob = wvb + (size_t)DIM * DIN;
  unsigned short* Qb  = wob + (size_t)DIM * DIM;
  unsigned short* Kb  = Qb  + (size_t)MROWS * DIM;
  unsigned short* Vb  = Kb  + (size_t)MROWS * DIM;
  unsigned short* Cb  = Vb  + (size_t)MROWS * DIM;

  cvt_f32_bf16<<<1024, 256, 0, stream>>>(x,  xb,  MROWS * DIN);
  cvt_f32_bf16<<<512,  256, 0, stream>>>(wq, wqb, DIM * DIN);
  cvt_f32_bf16<<<512,  256, 0, stream>>>(wk, wkb, DIM * DIN);
  cvt_f32_bf16<<<512,  256, 0, stream>>>(wv, wvb, DIM * DIN);
  cvt_f32_bf16<<<512,  256, 0, stream>>>(wo, wob, DIM * DIM);

  dim3 g(DIM / 128, MROWS / 128);   // (8, 32)
  gemm_bf16_nt<false><<<g, 256, 0, stream>>>(xb, wqb, nullptr, Qb, MROWS, DIM, DIN);
  gemm_bf16_nt<false><<<g, 256, 0, stream>>>(xb, wkb, nullptr, Kb, MROWS, DIM, DIN);
  gemm_bf16_nt<false><<<g, 256, 0, stream>>>(xb, wvb, nullptr, Vb, MROWS, DIM, DIN);

  dim3 fg(SEQ / 64, NHEADS, BATCH); // (32, 16, 2)
  flash_attn_causal<<<fg, 128, 0, stream>>>(Qb, Kb, Vb, Cb);

  gemm_bf16_nt<true><<<g, 256, 0, stream>>>(Cb, wob, out, nullptr, MROWS, DIM, DIM);
}
